// SparseDynamicHead_56075093017291
// MI455X (gfx1250) — compile-verified, matrix operand order checked by reading, not run on hardware
//
#include <hip/hip_runtime.h>
#include <hip/hip_bf16.h>

// ---------------------------------------------------------------------------
// SparseDynamicHead for MI455X (gfx1250, wave32, WMMA + TDM)
//   h = einsum('nkc,hkco->hno', gather(features, nbr), W3)   (5 heads)
//   BN(train stats) + ReLU + per-head 1x1 conv
// ---------------------------------------------------------------------------

typedef _Float16 v16h __attribute__((ext_vector_type(16)));
typedef _Float16 v8h  __attribute__((ext_vector_type(8)));
typedef float    v8f  __attribute__((ext_vector_type(8)));
typedef unsigned int u32x4 __attribute__((ext_vector_type(4)));
typedef int          i32x8 __attribute__((ext_vector_type(8)));
typedef int          i32x4 __attribute__((ext_vector_type(4)));

#define NHEADS 5
#define CIN    64
#define KKOFF  9
#define KTOT   (KKOFF * CIN)        /* 576 */
#define KB_N   (KTOT / 32)          /* 18 K-blocks of 32 */
#define CT_N   (CIN / 16)           /* 4 col tiles */
#define WAVES  8
#define B_FRAG_HALVES (KB_N * CT_N * 32 * 16)      /* 36864 halves = 72KB */
#define B_BYTES (B_FRAG_HALVES * 2)
#define RED_BYTES (WAVES * CT_N * 32 * 2 * 4)      /* 8KB */

// ---------------------------------------------------------------------------
// Kernel 1: repack W3 (fp32 [5,9,64,64]) into f16 B-fragments, in the per-lane
// v16h order of V_WMMA_F32_16X16X32_F16's B operand (32x16, lane = column):
//   Bpack[h][kb][ct][lane][0..15];  grp0 -> K {0..7,16..23}, grp1 -> {8..15,24..31}
// ---------------------------------------------------------------------------
__global__ void repack_w3_kernel(const float* __restrict__ W3,
                                 _Float16* __restrict__ bpack) {
    int u = blockIdx.x * blockDim.x + threadIdx.x;
    const int TOTAL = NHEADS * KB_N * CT_N * 32;
    if (u >= TOTAL) return;
    int lane = u & 31;
    int ct   = (u >> 5) & 3;
    int kb   = (u >> 7) % KB_N;
    int h    = (u >> 7) / KB_N;
    int grp  = lane >> 4;
    int col  = ct * 16 + (lane & 15);
    _Float16* dst = bpack + (size_t)u * 16;
#pragma unroll
    for (int i = 0; i < 16; ++i) {
        int klocal = (i < 8) ? ((grp ? 8 : 0) + i) : ((grp ? 24 : 16) + (i - 8));
        int K   = kb * 32 + klocal;   // global K in [0,576)
        int o   = K >> 6;             // 3x3 offset index
        int cin = K & 63;             // input channel
        float v = W3[(((size_t)h * KKOFF + o) * CIN + cin) * CIN + col];
        dst[i] = (_Float16)v;
    }
}

// ---------------------------------------------------------------------------
// Kernel 2: features fp32 -> f16, 8 elems / thread (b128 traffic)
// ---------------------------------------------------------------------------
__global__ void cast_feat_kernel(const float* __restrict__ f,
                                 _Float16* __restrict__ f16, int total8) {
    int i = blockIdx.x * blockDim.x + threadIdx.x;
    if (i >= total8) return;
    const float* p = f + (size_t)i * 8;
    v8h o;
#pragma unroll
    for (int t = 0; t < 8; ++t) o[t] = (_Float16)p[t];
    *(v8h*)(f16 + (size_t)i * 8) = o;
}

// ---------------------------------------------------------------------------
// Kernel 3: gather-GEMM with WMMA.  One wave = 16 rows x 64 cols, one head.
// Fully unrolled over the 9 conv offsets: 9 x 2 x 4 = 72 WMMAs per wave.
// B fragments (72KB) staged into dynamic LDS by the Tensor Data Mover
// (tensor_load_to_lds + s_wait_tensorcnt), falling back to a cooperative
// uint4 copy if the TDM builtin is unavailable.
// Deterministic per-block BN partials (sum, sumsq) via fixed-order LDS tree.
// ---------------------------------------------------------------------------
__global__ void gemm_kernel(const _Float16* __restrict__ F16,
                            const int* __restrict__ nbr,
                            const _Float16* __restrict__ bpack,
                            _Float16* __restrict__ HBUF,
                            float* __restrict__ PART,
                            int N, int NSTRIPS, int NBLK) {
    extern __shared__ char smem_raw[];
    _Float16* Bs  = (_Float16*)smem_raw;
    float*    red = (float*)(smem_raw + B_BYTES);

    const int tid  = threadIdx.x;
    const int wave = tid >> 5;
    const int lane = tid & 31;
    const int grp  = lane >> 4;
    const int nn   = lane & 15;
    const int head = blockIdx.y;
    const int bx   = blockIdx.x;

#if __has_builtin(__builtin_amdgcn_tensor_load_to_lds) && \
    __has_builtin(__builtin_amdgcn_s_wait_tensorcnt)
    // --- TDM: DMA this head's 72KB of B fragments into LDS (wave 0 issues).
    // D# per CDNA5 ISA: 1-D tile, 8-byte elements,
    // 9216 elements.  lds_addr = addr[31:0] of generic LDS pointer.
    if (wave == 0) {
        const unsigned long long ga =
            (unsigned long long)(bpack + (size_t)head * B_FRAG_HALVES);
        const unsigned int lds_lo = (unsigned int)(unsigned long long)Bs;
        const unsigned int ELEMS  = B_BYTES / 8;               // 9216
        u32x4 g0;
        g0[0] = 1u;                                            // count=1, user mode
        g0[1] = lds_lo;                                        // lds_addr
        g0[2] = (unsigned int)(ga & 0xffffffffu);              // global_addr lo
        g0[3] = (unsigned int)((ga >> 32) & 0x01ffffffu) | (2u << 30); // hi | type=2
        i32x8 g1;
        g1[0] = (int)(3u << 16);                               // mask=0, data_size=8B
        g1[1] = (int)((ELEMS & 0xffffu) << 16);                // tensor_dim0 lo16
        g1[2] = (int)(((ELEMS >> 16) & 0xffffu) | (1u << 16)); // dim0 hi | tensor_dim1=1
        g1[3] = (int)(ELEMS << 16);                            // tile_dim0 = 9216
        g1[4] = 0;                                             // tile_dim1/2 unused
        g1[5] = (int)ELEMS;                                    // tensor_dim0_stride
        g1[6] = 0;
        g1[7] = 0;
        i32x4 g2 = {0, 0, 0, 0};
        i32x4 g3 = {0, 0, 0, 0};
#if __clang_major__ >= 23
        i32x8 g1b = {0, 0, 0, 0, 0, 0, 0, 0};
        __builtin_amdgcn_tensor_load_to_lds(g0, g1, g2, g3, g1b, 0);
#else
        __builtin_amdgcn_tensor_load_to_lds(g0, g1, g2, g3, 0);
#endif
        __builtin_amdgcn_s_wait_tensorcnt(0);
    }
#else
    // --- fallback: cooperative stage of B fragments into LDS (16B chunks)
    {
        const uint4* src = (const uint4*)(bpack + (size_t)head * B_FRAG_HALVES);
        uint4* dst = (uint4*)Bs;
        const int CHUNKS = B_BYTES / 16;                 // 4608
        for (int i = tid; i < CHUNKS; i += blockDim.x) dst[i] = src[i];
    }
#endif
    __syncthreads();

    const int strip  = bx * WAVES + wave;
    const bool active = strip < NSTRIPS;

    v8f acc[CT_N] = {};

    if (active) {
        const int row = strip * 16 + nn;                 // this lane's M row
        int nidx[KKOFF];
#pragma unroll
        for (int o = 0; o < KKOFF; ++o) {
            nidx[o] = nbr[(size_t)row * KKOFF + o];
            if (nidx[o] >= 0)
                __builtin_prefetch(F16 + (size_t)nidx[o] * CIN, 0, 0);
        }

#pragma unroll
        for (int o = 0; o < KKOFF; ++o) {
            // --- gather the full 64-ch neighbor row once: 2 A fragments
            v16h a0 = {}, a1 = {};
            const int idx = nidx[o];
            if (idx >= 0) {
                const _Float16* p = F16 + (size_t)idx * CIN + (grp ? 8 : 0);
                v8h x0 = *(const v8h*)p;          // K {0..7} / {8..15}
                v8h x1 = *(const v8h*)(p + 16);   // K {16..23} / {24..31}
                v8h x2 = *(const v8h*)(p + 32);   // +32 channel half
                v8h x3 = *(const v8h*)(p + 48);
                a0 = __builtin_shufflevector(x0, x1,
                        0, 1, 2, 3, 4, 5, 6, 7, 8, 9, 10, 11, 12, 13, 14, 15);
                a1 = __builtin_shufflevector(x2, x3,
                        0, 1, 2, 3, 4, 5, 6, 7, 8, 9, 10, 11, 12, 13, 14, 15);
            }
            const int kb0 = o * 2, kb1 = o * 2 + 1;
#pragma unroll
            for (int ct = 0; ct < CT_N; ++ct) {
                const v16h b = *(const v16h*)
                    (Bs + ((size_t)(kb0 * CT_N + ct) * 32 + lane) * 16);
                acc[ct] = __builtin_amdgcn_wmma_f32_16x16x32_f16(
                    false, a0, false, b, (short)0, acc[ct], false, false);
            }
#pragma unroll
            for (int ct = 0; ct < CT_N; ++ct) {
                const v16h b = *(const v16h*)
                    (Bs + ((size_t)(kb1 * CT_N + ct) * 32 + lane) * 16);
                acc[ct] = __builtin_amdgcn_wmma_f32_16x16x32_f16(
                    false, a1, false, b, (short)0, acc[ct], false, false);
            }
        }

        // --- store pre-BN activations h[head][row][col] as f16 (halves HBM)
#pragma unroll
        for (int ct = 0; ct < CT_N; ++ct) {
            const int col = ct * 16 + nn;
#pragma unroll
            for (int j = 0; j < 8; ++j) {
                const int grow = strip * 16 + grp * 8 + j;
                HBUF[((size_t)head * N + grow) * CIN + col] = (_Float16)acc[ct][j];
            }
        }
    }

    // --- per-lane BN partials from fp32 accumulators (zeros if inactive)
#pragma unroll
    for (int ct = 0; ct < CT_N; ++ct) {
        float s = 0.f, q = 0.f;
#pragma unroll
        for (int j = 0; j < 8; ++j) {
            const float v = acc[ct][j];
            s += v;
            q += v * v;
        }
        const int slot = ((wave * CT_N + ct) * 32 + lane) * 2;
        red[slot]     = s;
        red[slot + 1] = q;
    }
    __syncthreads();

    // --- deterministic block reduce: thread ch<64 folds 8 waves x 2 groups
    if (tid < CIN) {
        const int ch = tid;
        const int ct = ch >> 4;
        const int nl = ch & 15;
        float s = 0.f, q = 0.f;
        for (int w = 0; w < WAVES; ++w)
            for (int g = 0; g < 2; ++g) {
                const int slot = ((w * CT_N + ct) * 32 + g * 16 + nl) * 2;
                s += red[slot];
                q += red[slot + 1];
            }
        float* pp = PART + ((size_t)head * NBLK + bx) * 128;
        pp[ch]      = s;
        pp[ch + 64] = q;
    }
}

// ---------------------------------------------------------------------------
// Kernel 4: fold per-block partials -> mean / rstd per (head, channel)
// ---------------------------------------------------------------------------
__global__ void stats_kernel(const float* __restrict__ PART,
                             float* __restrict__ STATS, int N, int NBLK) {
    int t = blockIdx.x * blockDim.x + threadIdx.x;
    if (t >= NHEADS * CIN) return;
    const int hd = t >> 6, ch = t & 63;
    float S = 0.f, Q = 0.f;
    for (int b = 0; b < NBLK; ++b) {
        const float* pp = PART + ((size_t)hd * NBLK + b) * 128;
        S += pp[ch];
        Q += pp[ch + 64];
    }
    const float inv  = 1.0f / (float)N;
    const float mean = S * inv;
    float var = Q * inv - mean * mean;
    var = fmaxf(var, 0.f);
    STATS[hd * 128 + ch]      = mean;
    STATS[hd * 128 + 64 + ch] = __frsqrt_rn(var + 1e-5f);
}

// ---------------------------------------------------------------------------
// Kernel 5: BN + ReLU + per-head 1x1 conv, tuple-concatenated output
//   out layout: heatmap[N,3] | center[N,2] | center_z[N,1] | dim[N,3] | rot[N,2]
// ---------------------------------------------------------------------------
__global__ void head_kernel(const _Float16* __restrict__ HBUF,
                            const float* __restrict__ STATS,
                            const float* __restrict__ gamma,
                            const float* __restrict__ beta,
                            const float* __restrict__ W0, const float* __restrict__ B0,
                            const float* __restrict__ W1, const float* __restrict__ B1,
                            const float* __restrict__ W2, const float* __restrict__ B2,
                            const float* __restrict__ W3p, const float* __restrict__ B3,
                            const float* __restrict__ W4, const float* __restrict__ B4,
                            float* __restrict__ out, int N) {
    int n = blockIdx.x * blockDim.x + threadIdx.x;
    if (n >= N) return;
    const float* Ws[NHEADS] = {W0, W1, W2, W3p, W4};
    const float* Bs[NHEADS] = {B0, B1, B2, B3, B4};
    const int ocs[NHEADS]   = {3, 2, 1, 3, 2};
    const int cum[NHEADS]   = {0, 3, 5, 6, 9};
#pragma unroll
    for (int hd = 0; hd < NHEADS; ++hd) {
        const int oc = ocs[hd];
        float acc[3];
        for (int j = 0; j < 3; ++j) acc[j] = (j < oc) ? Bs[hd][j] : 0.f;
        const _Float16* hrow = HBUF + ((size_t)hd * N + n) * CIN;
        const float* st = STATS + hd * 128;
        const float* gm = gamma + hd * CIN;
        const float* bt = beta + hd * CIN;
        const float* Wp = Ws[hd];
#pragma unroll
        for (int c8 = 0; c8 < 8; ++c8) {
            const v8h hv = *(const v8h*)(hrow + c8 * 8);   // b128 load
#pragma unroll
            for (int t = 0; t < 8; ++t) {
                const int c = c8 * 8 + t;
                float v = ((float)hv[t] - st[c]) * st[64 + c] * gm[c] + bt[c];
                v = fmaxf(v, 0.f);
                for (int j = 0; j < oc; ++j) acc[j] += v * Wp[c * oc + j];
            }
        }
        float* op = out + (size_t)cum[hd] * N + (size_t)n * oc;
        for (int j = 0; j < oc; ++j) op[j] = acc[j];
    }
}

// ---------------------------------------------------------------------------
extern "C" void kernel_launch(void* const* d_in, const int* in_sizes, int n_in,
                              void* d_out, int out_size, void* d_ws, size_t ws_size,
                              hipStream_t stream) {
    const float* features = (const float*)d_in[0];
    const int*   nbr      = (const int*)d_in[1];
    const float* W3       = (const float*)d_in[2];
    const float* gamma    = (const float*)d_in[3];
    const float* beta     = (const float*)d_in[4];
    const float* W1s[NHEADS], *b1s[NHEADS];
    for (int i = 0; i < NHEADS; ++i) {
        W1s[i] = (const float*)d_in[5 + 2 * i];
        b1s[i] = (const float*)d_in[6 + 2 * i];
    }

    const int N       = in_sizes[0] / CIN;      // 150000
    const int NSTRIPS = N / 16;                 // 9375 (N % 16 == 0)
    const int NBLK    = (NSTRIPS + WAVES - 1) / WAVES;

    // workspace carve-up (all section sizes are multiples of 256B)
    char* ws = (char*)d_ws;
    size_t off = 0;
    _Float16* HBUF = (_Float16*)(ws + off); off += (size_t)NHEADS * N * CIN * 2;
    float* PART = (float*)(ws + off);  off += (size_t)NHEADS * NBLK * 128 * 4;
    float* STATS = (float*)(ws + off); off += (size_t)NHEADS * 128 * 4;
    _Float16* BPACK = (_Float16*)(ws + off);
    off += (size_t)NHEADS * B_FRAG_HALVES * 2;
    _Float16* F16 = (_Float16*)(ws + off);

    // 1) weight repack into WMMA B-fragment layout
    {
        const int total = NHEADS * KB_N * CT_N * 32;   // 11520
        repack_w3_kernel<<<(total + 255) / 256, 256, 0, stream>>>(W3, BPACK);
    }
    // 2) features -> f16 (vectorized x8)
    {
        const int total8 = (N * CIN) / 8;
        cast_feat_kernel<<<(total8 + 255) / 256, 256, 0, stream>>>(features, F16, total8);
    }
    // 3) WMMA gather-GEMM + BN partials (B staged via TDM)
    {
        dim3 grid(NBLK, NHEADS);
        gemm_kernel<<<grid, 256, B_BYTES + RED_BYTES, stream>>>(
            F16, nbr, BPACK, HBUF, PART, N, NSTRIPS, NBLK);
    }
    // 4) BN stats
    stats_kernel<<<2, 256, 0, stream>>>(PART, STATS, N, NBLK);
    // 5) BN + ReLU + 1x1 heads
    head_kernel<<<(N + 255) / 256, 256, 0, stream>>>(
        HBUF, STATS, gamma, beta,
        W1s[0], b1s[0], W1s[1], b1s[1], W1s[2], b1s[2],
        W1s[3], b1s[3], W1s[4], b1s[4], (float*)d_out, N);
}
